// RateRNN_10728828305920
// MI455X (gfx1250) — matrix operand order repaired and checked by reading, not compiled, not run on hardware
//
#include <hip/hip_runtime.h>
#include <hip/hip_bf16.h>

// ---------------------------------------------------------------------------
// RateRNN on MI455X (gfx1250): f16 WMMA (f32 accumulate) for all GEMMs.
//   h_t  = (1-a) h_{t-1} + a (sigmoid(h_{t-1}) W_hh^T + x_t W_ih^T) + 0.01 n_t
//   out  = sigmoid(hidden) W_ho^T + b_ho
// Key idea vs previous round: sigmoid is computed ONCE per element, fused in
// the step-kernel epilogue, and stored as f16; GEMM A-fragments are then pure
// f16 dword loads (no per-tile VALU transcendentals, no IEEE divides).
// ---------------------------------------------------------------------------

typedef __attribute__((ext_vector_type(16))) _Float16 v16h;
typedef __attribute__((ext_vector_type(8)))  float    v8f;

namespace {
constexpr int Bn = 64;    // batch
constexpr int Sn = 1024;  // sequence
constexpr int In = 512;   // input dim
constexpr int Hn = 2048;  // hidden dim
constexpr int On = 512;   // output dim
constexpr float A_BLEND = 0.1f;  // dt/tau
}

// Fast sigmoid: v_mul + v_exp_f32 + v_add + v_rcp_f32 (no IEEE divide).
__device__ __forceinline__ float fast_sigmoid(float x) {
    return __builtin_amdgcn_rcpf(1.0f + __expf(-x));
}

// ---------------- WMMA fragment loaders (CDNA5 16-bit layouts) -------------
// A-fragment (16x32 f16): lane L: m=L&15, half=L>>4; VGPR v holds
// K = (v<4 ? 2v : 2v+8) + 8*half, +1.

__device__ __forceinline__ v16h load_a_f16(const _Float16* __restrict__ src,
                                           int row, long ld, int kbase, int lane) {
    const int m    = lane & 15;
    const int half = lane >> 4;
    const _Float16* p = src + (size_t)(row + m) * (size_t)ld + kbase + half * 8;
    v16h a;
#pragma unroll
    for (int v = 0; v < 8; ++v) {
        const int k0 = (v < 4) ? (2 * v) : (2 * v + 8);
        union { unsigned u; _Float16 h[2]; } c;
        c.u = *(const unsigned*)(p + k0);
        a[2 * v]     = c.h[0];
        a[2 * v + 1] = c.h[1];
    }
    return a;
}

template <bool SIG>
__device__ __forceinline__ v16h load_a_f32(const float* __restrict__ src,
                                           int row, long ld, int kbase, int lane) {
    const int m    = lane & 15;
    const int half = lane >> 4;
    const float* p = src + (size_t)(row + m) * (size_t)ld + kbase + half * 8;
    v16h a;
#pragma unroll
    for (int v = 0; v < 8; ++v) {
        const int k0 = (v < 4) ? (2 * v) : (2 * v + 8);
        float2 f = *(const float2*)(p + k0);
        float x0 = SIG ? fast_sigmoid(f.x) : f.x;
        float x1 = SIG ? fast_sigmoid(f.y) : f.y;
        a[2 * v]     = (_Float16)x0;
        a[2 * v + 1] = (_Float16)x1;
    }
    return a;
}

// B-fragment (32x16 f16): B[k][n] = W[col+n][kbase+k], W row-major [N,K].
// Lane L: n=L&15, half=L>>4; VGPR v holds K = (v<4 ? 2v : 2v+8) + 8*half, +1 (dword load).
__device__ __forceinline__ v16h load_b_w16(const _Float16* __restrict__ W,
                                           int col, int ld, int kbase, int lane) {
    const int n    = lane & 15;
    const int half = lane >> 4;
    const _Float16* p = W + (size_t)(col + n) * (size_t)ld + kbase + half * 8;
    v16h b;
#pragma unroll
    for (int v = 0; v < 8; ++v) {
        union { unsigned u; _Float16 h[2]; } c;
        c.u = *(const unsigned*)(p + ((v < 4) ? (2 * v) : (2 * v + 8)));
        b[2 * v]     = c.h[0];
        b[2 * v + 1] = c.h[1];
    }
    return b;
}

__device__ __forceinline__ v8f wmma_f16(v16h a, v16h b, v8f c) {
    return __builtin_amdgcn_wmma_f32_16x16x32_f16(
        /*neg_a=*/false, a, /*neg_b=*/false, b,
        /*c_mod=*/(short)0, c, /*reuse_a=*/false, /*reuse_b=*/false);
}

// ---------------- elementwise prep kernels ---------------------------------

__global__ __launch_bounds__(256)
void cvt_f32_f16_kernel(const float* __restrict__ src, _Float16* __restrict__ dst,
                        int n4) {
    int i = blockIdx.x * blockDim.x + threadIdx.x;
    if (i < n4) {
        float4 f = ((const float4*)src)[i];
        union { _Float16 h[4]; unsigned long long u; } pk;
        pk.h[0] = (_Float16)f.x; pk.h[1] = (_Float16)f.y;
        pk.h[2] = (_Float16)f.z; pk.h[3] = (_Float16)f.w;
        ((unsigned long long*)dst)[i] = pk.u;
    }
}

__global__ __launch_bounds__(256)
void sigmoid_f16_kernel(const float* __restrict__ src, _Float16* __restrict__ dst,
                        int n) {
    int i = blockIdx.x * blockDim.x + threadIdx.x;
    if (i < n) dst[i] = (_Float16)fast_sigmoid(src[i]);
}

// ---------------- recurrence step ------------------------------------------
// Output [64 x 2048] = 4 x 128 tiles of 16x16; each wave owns TWO column
// tiles (one A-fragment feeds two WMMAs). 256 waves -> 32 blocks x 8 waves.
// A (recurrent) = s_prev (f16, sigmoid already applied). A (input) = x f16
// (pre-converted) or f32 on the fly. Epilogue: Euler blend + noise, writes
// h_next (f32), hidden slab (f32, into d_out), sigmoid(h_new) -> s_next f16
// (+ optional s_hidden f16 slab for the output projection).
template <bool XF16, bool WRITE_SH>
__global__ __launch_bounds__(256)
void rnn_step_kernel(const float* __restrict__ h_prev,      // [B,H] f32
                     float* __restrict__ h_next,            // [B,H] f32
                     const _Float16* __restrict__ s_prev,   // [B,H] f16
                     _Float16* __restrict__ s_next,         // [B,H] f16
                     const float* __restrict__ x_f32,       // [B,S,I] f32
                     const _Float16* __restrict__ x_f16,    // [B,S,I] f16
                     const float* __restrict__ noise,       // [S,B,H] f32
                     float* __restrict__ hidden,            // [B,S,H] f32
                     _Float16* __restrict__ s_hidden,       // [B,S,H] f16
                     const _Float16* __restrict__ Whh,      // [H,H] f16
                     const _Float16* __restrict__ Wih,      // [H,I] f16
                     int t) {
    const int lane = threadIdx.x & 31;
    const int wid  = (blockIdx.x * blockDim.x + threadIdx.x) >> 5;  // 0..255
    const int row  = (wid >> 6) * 16;        // row tile in [0,4)
    const int col0 = (wid & 63) * 32;        // pair of column tiles
    const int col1 = col0 + 16;

    v8f acc0 = {}, acc1 = {};
    // Recurrent term: A = sigmoid(h_{t-1}) (f16), B = Whh^T.
#pragma unroll 4
    for (int kk = 0; kk < Hn; kk += 32) {
        v16h a  = load_a_f16(s_prev, row, Hn, kk, lane);
        v16h b0 = load_b_w16(Whh, col0, Hn, kk, lane);
        v16h b1 = load_b_w16(Whh, col1, Hn, kk, lane);
        acc0 = wmma_f16(a, b0, acc0);
        acc1 = wmma_f16(a, b1, acc1);
    }
    // Input term: A = x[:, t, :] (row stride S*I), B = Wih^T.
#pragma unroll 4
    for (int kk = 0; kk < In; kk += 32) {
        v16h a;
        if (XF16) a = load_a_f16(x_f16 + (size_t)t * In, row, (long)Sn * In, kk, lane);
        else      a = load_a_f32<false>(x_f32 + (size_t)t * In, row, (long)Sn * In, kk, lane);
        v16h b0 = load_b_w16(Wih, col0, In, kk, lane);
        v16h b1 = load_b_w16(Wih, col1, In, kk, lane);
        acc0 = wmma_f16(a, b0, acc0);
        acc1 = wmma_f16(a, b1, acc1);
    }
    // Epilogue. C/D layout: lane L: n = L&15; VGPR r holds M = r + 8*(L>>4).
    const int nlo  = lane & 15;
    const int mofs = (lane >> 4) * 8;
#pragma unroll
    for (int tile = 0; tile < 2; ++tile) {
        const v8f& acc = tile ? acc1 : acc0;
        const int n_g = (tile ? col1 : col0) + nlo;
#pragma unroll
        for (int r = 0; r < 8; ++r) {
            const int m_g = row + r + mofs;                 // batch index
            const size_t hidx = (size_t)m_g * Hn + n_g;
            const size_t oidx = ((size_t)m_g * Sn + t) * Hn + n_g;
            const float hp = h_prev[hidx];
            const float nz = noise[((size_t)t * Bn + m_g) * Hn + n_g];
            const float hn = (1.0f - A_BLEND) * hp + A_BLEND * acc[r] + 0.01f * nz;
            h_next[hidx] = hn;
            hidden[oidx] = hn;
            const _Float16 sg = (_Float16)fast_sigmoid(hn);
            s_next[hidx] = sg;
            if (WRITE_SH) s_hidden[oidx] = sg;
        }
    }
}

// ---------------- output projection ----------------------------------------
// out = sigmoid(hidden) @ W_ho^T + b_ho : [65536 x 2048] x [2048 x 512].
// Two column tiles per wave: 4096 x 16 wave-jobs = 65536 waves -> 8192 blocks.
template <bool SHF16>
__global__ __launch_bounds__(256)
void out_proj_kernel(const float* __restrict__ hidden,     // [B*S,H] f32
                     const _Float16* __restrict__ s_hidden,// [B*S,H] f16
                     const float* __restrict__ b_ho,       // [O] f32
                     const _Float16* __restrict__ Who,     // [O,H] f16
                     float* __restrict__ out) {            // [B*S,O] f32
    const int lane = threadIdx.x & 31;
    const int wid  = (blockIdx.x * blockDim.x + threadIdx.x) >> 5;
    const int row  = (wid >> 4) * 16;   // row tile in [0,4096)
    const int col0 = (wid & 15) * 32;   // pair of column tiles
    const int col1 = col0 + 16;

    v8f acc0 = {}, acc1 = {};
#pragma unroll 4
    for (int kk = 0; kk < Hn; kk += 32) {
        v16h a;
        if (SHF16) a = load_a_f16(s_hidden, row, Hn, kk, lane);
        else       a = load_a_f32<true>(hidden, row, Hn, kk, lane);
        v16h b0 = load_b_w16(Who, col0, Hn, kk, lane);
        v16h b1 = load_b_w16(Who, col1, Hn, kk, lane);
        acc0 = wmma_f16(a, b0, acc0);
        acc1 = wmma_f16(a, b1, acc1);
    }
    const int nlo  = lane & 15;
    const int mofs = (lane >> 4) * 8;
#pragma unroll
    for (int tile = 0; tile < 2; ++tile) {
        const v8f& acc = tile ? acc1 : acc0;
        const int n_g = (tile ? col1 : col0) + nlo;
        const float bias = b_ho[n_g];
#pragma unroll
        for (int r = 0; r < 8; ++r) {
            const int m_g = row + r + mofs;
            out[(size_t)m_g * On + n_g] = acc[r] + bias;
        }
    }
}

// ---------------- host launcher --------------------------------------------

extern "C" void kernel_launch(void* const* d_in, const int* in_sizes, int n_in,
                              void* d_out, int out_size, void* d_ws, size_t ws_size,
                              hipStream_t stream) {
    (void)in_sizes; (void)n_in; (void)out_size;
    const float* x     = (const float*)d_in[0];  // [B,S,I]
    const float* h0    = (const float*)d_in[1];  // [B,1,H]
    const float* W_ih  = (const float*)d_in[2];  // [H,I]
    const float* W_hh  = (const float*)d_in[3];  // [H,H]
    const float* W_ho  = (const float*)d_in[4];  // [O,H]
    const float* b_ho  = (const float*)d_in[5];  // [O]
    const float* noise = (const float*)d_in[6];  // [S,B,1,H]

    // Tuple return (out, hidden): out first, hidden second, flat in d_out.
    float* out    = (float*)d_out;                         // [B,S,O]
    float* hidden = (float*)d_out + (size_t)Bn * Sn * On;  // [B,S,H]

    // Workspace carve (256B-aligned slices).
    size_t off = 0;
    char* base = (char*)d_ws;
    auto carve = [&](size_t bytes) -> void* {
        void* p = base + off;
        off += (bytes + 255) & ~(size_t)255;
        return p;
    };
    _Float16* Whh_h = (_Float16*)carve(2ull * Hn * Hn);
    _Float16* Wih_h = (_Float16*)carve(2ull * Hn * In);
    _Float16* Who_h = (_Float16*)carve(2ull * On * Hn);
    float*    hbuf0 = (float*)   carve(4ull * Bn * Hn);
    float*    hbuf1 = (float*)   carve(4ull * Bn * Hn);
    _Float16* sbuf0 = (_Float16*)carve(2ull * Bn * Hn);
    _Float16* sbuf1 = (_Float16*)carve(2ull * Bn * Hn);
    const size_t tierB_need = off;
    _Float16* x_h  = (_Float16*)carve(2ull * Bn * Sn * In);        // 64 MB
    _Float16* s_hid = (_Float16*)carve(2ull * (size_t)Bn * Sn * Hn); // 256 MB
    const size_t tierA_need = off;

    const bool big = (ws_size >= tierA_need);
    (void)tierB_need;

    // 1) Weight conversion f32 -> f16 (all counts divisible by 4).
    {
        int n4;
        n4 = (Hn * Hn) / 4;
        cvt_f32_f16_kernel<<<(n4 + 255) / 256, 256, 0, stream>>>(W_hh, Whh_h, n4);
        n4 = (Hn * In) / 4;
        cvt_f32_f16_kernel<<<(n4 + 255) / 256, 256, 0, stream>>>(W_ih, Wih_h, n4);
        n4 = (On * Hn) / 4;
        cvt_f32_f16_kernel<<<(n4 + 255) / 256, 256, 0, stream>>>(W_ho, Who_h, n4);
    }
    // 2) Pre-convert x to f16 (big tier only) and seed s0 = sigmoid(h0).
    if (big) {
        int n4 = (int)(((size_t)Bn * Sn * In) / 4);
        cvt_f32_f16_kernel<<<(n4 + 255) / 256, 256, 0, stream>>>(x, x_h, n4);
    }
    {
        int n = Bn * Hn;
        sigmoid_f16_kernel<<<(n + 255) / 256, 256, 0, stream>>>(h0, sbuf0, n);
    }

    // 3) Sequential scan: 1024 launches, h/s ping-pong through workspace.
    //    Step t reads (h_prev, s_prev) and writes (h_next, s_next):
    //    read buf index (t&1)==0 -> buf0, write -> buf1, alternating.
    const float*    hp = h0;
    const _Float16* sp = sbuf0;
    for (int t = 0; t < Sn; ++t) {
        float*    hn = (t & 1) ? hbuf0 : hbuf1;   // write alternates
        _Float16* sn = (t & 1) ? sbuf0 : sbuf1;
        if (big) {
            rnn_step_kernel<true, true><<<32, 256, 0, stream>>>(
                hp, hn, sp, sn, x, x_h, noise, hidden, s_hid, Whh_h, Wih_h, t);
        } else {
            rnn_step_kernel<false, false><<<32, 256, 0, stream>>>(
                hp, hn, sp, sn, x, x_h, noise, hidden, s_hid, Whh_h, Wih_h, t);
        }
        hp = hn;
        sp = sn;
    }

    // 4) Output projection: 65536 waves -> 8192 blocks x 256 threads.
    if (big) {
        out_proj_kernel<true><<<8192, 256, 0, stream>>>(hidden, s_hid, b_ho,
                                                        Who_h, out);
    } else {
        out_proj_kernel<false><<<8192, 256, 0, stream>>>(hidden, s_hid, b_ho,
                                                         Who_h, out);
    }
}
